// PCRNN_19928648253953
// MI455X (gfx1250) — matrix operand order repaired and checked by reading, not compiled, not run on hardware
//
#include <hip/hip_runtime.h>
#include <stdint.h>

// Problem sizes
#define Bsz 128
#define Tn  512
#define In  512
#define Hn  2048
#define On  512
#define KC  (In + Hn)   // 2560 combined K for fused enc+rec GEMM
#define NBLK 32         // persistent blocks (each: 256 threads = 8 wave32)

#ifndef __has_builtin
#define __has_builtin(x) 0
#endif

#if __has_builtin(__builtin_amdgcn_global_load_async_to_lds_b128)
#define HAS_ASYNC_LDS 1
#else
#define HAS_ASYNC_LDS 0
#endif

typedef __attribute__((ext_vector_type(16))) __bf16 v16bf;
typedef __attribute__((ext_vector_type(8)))  float  v8f;
typedef int v4i_vs __attribute__((vector_size(16)));   // matches builtin's param type

__device__ __forceinline__ unsigned short f2bf(float x) {
  unsigned int u = __float_as_uint(x);
  u += 0x7fffu + ((u >> 16) & 1u);      // round-to-nearest-even
  return (unsigned short)(u >> 16);
}

union FragBF {
  uint4 q[2];
  v16bf v;
};

// 16 bytes global -> LDS. CDNA5 async-to-LDS (ASYNCcnt), fallback = manual.
__device__ __forceinline__ void stage16(const unsigned short* g, unsigned short* l) {
#if HAS_ASYNC_LDS
  __builtin_amdgcn_global_load_async_to_lds_b128(
      (__attribute__((address_space(1))) v4i_vs*)(unsigned short*)g,
      (__attribute__((address_space(3))) v4i_vs*)l, 0, 0);
#else
  *(uint4*)l = *(const uint4*)g;
#endif
}

__device__ __forceinline__ void wait_async0() {
#if HAS_ASYNC_LDS
#if __has_builtin(__builtin_amdgcn_s_wait_asynccnt)
  __builtin_amdgcn_s_wait_asynccnt(0);
#else
  asm volatile("s_wait_asynccnt 0" ::: "memory");
#endif
#endif
}

// A fragment (16x32 bf16): lane L<16 -> row m0+L, K = {0..7, 16..23};
// lane L>=16 -> row m0+L-16, K = {8..15, 24..31}. Two b128 loads per lane.
__device__ __forceinline__ v16bf load_a_frag(const unsigned short* p0, size_t rowstride) {
  const int lane = threadIdx.x & 31;
  const unsigned short* p = p0 + (size_t)(lane & 15) * rowstride + (size_t)((lane >> 4) * 8);
  FragBF f;
  f.q[0] = *(const uint4*)(p);
  f.q[1] = *(const uint4*)(p + 16);
  return f.v;
}

// B fragment (32x16 bf16) from an LDS tile stored [k][n]-major:
// lane L holds column n=noff+(L&15), K = {8h..8h+7, 16+8h..16+8h+7}, h=L>>4.
__device__ __forceinline__ v16bf load_b_frag(const unsigned short* bt, int stride, int noff) {
  const int lane = threadIdx.x & 31;
  const unsigned short* p = bt + (lane >> 4) * 8 * stride + noff + (lane & 15);
  union { unsigned short s[16]; v16bf v; } f;
#pragma unroll
  for (int s = 0; s < 8; ++s) {
    f.s[s]     = p[s * stride];
    f.s[8 + s] = p[(16 + s) * stride];
  }
  return f.v;
}

#define WMMA_BF16(A, Bf, C) \
  __builtin_amdgcn_wmma_f32_16x16x32_bf16(false, (A), false, (Bf), (short)0, (C), false, false)

// Decoder GEMM for one timestep: out[td] slice [128 x 16] = h(bf16) @ wdec + dec_b
// K staged 64 at a time (LDS tile [64][24], double buffered); last chunk peeled.
__device__ __forceinline__ void dec_step(int td, const unsigned short* hbuf,
                                         const unsigned short* wdec, const float* dec_b,
                                         float* out, unsigned short* BtD /* [2][64*24] */) {
  const int tid  = threadIdx.x;
  const int wave = tid >> 5;
  const int lane = tid & 31;
  const int m0   = wave * 16;
  const int nbD  = blockIdx.x * 16;

  v8f acc = {};

  if (tid < 128) {  // stage chunk 0: 64(K) x 16(N) bf16 tile
    int row = tid >> 1, col = (tid & 1) * 8;
    stage16(wdec + (size_t)row * On + nbD + col, BtD + row * 24 + col);
  }
  wait_async0();
  __syncthreads();

  const unsigned short* arow = hbuf + (size_t)m0 * Hn;
  v16bf aF0 = load_a_frag(arow + 0, Hn);
  v16bf aF1 = load_a_frag(arow + 32, Hn);

  const int NCH = Hn / 64;  // 32
#pragma unroll 2
  for (int c = 0; c < NCH - 1; ++c) {
    const int cur = c & 1;
    if (tid < 128) {  // unconditional (in steady state) async prefetch of next tile
      int row = tid >> 1, col = (tid & 1) * 8;
      stage16(wdec + (size_t)((c + 1) * 64 + row) * On + nbD + col,
              BtD + (cur ^ 1) * (64 * 24) + row * 24 + col);
    }
    v16bf aN0 = load_a_frag(arow + (size_t)(c + 1) * 64, Hn);
    v16bf aN1 = load_a_frag(arow + (size_t)(c + 1) * 64 + 32, Hn);

    const unsigned short* bt = BtD + cur * (64 * 24);
    v16bf b0 = load_b_frag(bt, 24, 0);
    acc = WMMA_BF16(aF0, b0, acc);
    v16bf b1 = load_b_frag(bt + 32 * 24, 24, 0);
    acc = WMMA_BF16(aF1, b1, acc);
    aF0 = aN0;
    aF1 = aN1;
    wait_async0();
    __syncthreads();
  }
  {  // peeled last chunk: buffer parity (NCH-1)&1
    const unsigned short* bt = BtD + ((NCH - 1) & 1) * (64 * 24);
    v16bf b0 = load_b_frag(bt, 24, 0);
    acc = WMMA_BF16(aF0, b0, acc);
    v16bf b1 = load_b_frag(bt + 32 * 24, 24, 0);
    acc = WMMA_BF16(aF1, b1, acc);
    __syncthreads();
  }

  const int n  = nbD + (lane & 15);
  const int mb = m0 + 8 * (lane >> 4);
  const float db = dec_b[n];
#pragma unroll
  for (int r = 0; r < 8; ++r) {
    out[((size_t)td * Bsz + (size_t)(mb + r)) * On + n] = acc[r] + db;
  }
}

__global__ __launch_bounds__(256, 1) void rnn_persistent(
    const unsigned short* __restrict__ xb,     // [T*B, I] bf16, time-major
    const unsigned short* __restrict__ wcomb,  // [KC, H] bf16 (enc_w^T ; a*rec_w^T)
    const unsigned short* __restrict__ wdec,   // [H, O] bf16 (dec_w^T)
    const float* __restrict__ bias_c,          // [H] enc_b + a*rec_b
    const float* __restrict__ dec_b,           // [O]
    const float* __restrict__ dt,              // [1]
    float* __restrict__ hf,                    // [2][B*H] fp32 (step parity)
    unsigned short* __restrict__ hb,           // [2][B*H] bf16 (step parity)
    float* __restrict__ out,                   // [T*B*O] then [B*H]
    unsigned int* __restrict__ bar)            // [T] per-step arrival counters
{
  __shared__ unsigned short BtR[2 * 64 * 72];  // rec B tiles [k][n] (+pad), K=64 stage
  __shared__ unsigned short BtD[2 * 64 * 24];  // dec B tiles

  const int tid  = threadIdx.x;
  const int wave = tid >> 5;
  const int lane = tid & 31;
  const int m0   = wave * 16;            // wave's 16-row M strip of the 128 batch rows
  const int nb   = blockIdx.x * 64;      // block's 64-col N slice of H

  const float sg  = 1.0f / (1.0f + __expf(-dt[0]));
  const float omg = 1.0f - sg;

#pragma unroll 1
  for (int t = 0; t < Tn; ++t) {
    const int curp = t & 1;
    const unsigned short* hb_cur = hb + (size_t)curp * (Bsz * Hn);
    unsigned short*       hb_nxt = hb + (size_t)(curp ^ 1) * (Bsz * Hn);
    const float*          hf_cur = hf + (size_t)curp * (Bsz * Hn);
    float*                hf_nxt = hf + (size_t)(curp ^ 1) * (Bsz * Hn);

    // y_{t-1} = h_t @ dec_w^T + dec_b  (overlaps with this step's rec GEMM inputs)
    if (t > 0) dec_step(t - 1, hb_cur, wdec, dec_b, out, BtD);

    // ---- fused rec GEMM: u[128 x (nb..nb+64)] = [x_t, h_t] @ wcomb, K = 2560 ----
    v8f a0 = {}, a1 = {}, a2 = {}, a3 = {};
    {
      int row = tid >> 3, col = (tid & 7) * 8;   // stage chunk 0: 64x64 tile
      const unsigned short* s0 = wcomb + (size_t)row * Hn + nb + col;
      stage16(s0, BtR + row * 72 + col);
      stage16(s0 + (size_t)32 * Hn, BtR + (row + 32) * 72 + col);
    }
    wait_async0();
    __syncthreads();

    const unsigned short* xrow = xb + (size_t)t * Bsz * In;
    v16bf aF0 = load_a_frag(xrow + (size_t)m0 * In + 0, In);
    v16bf aF1 = load_a_frag(xrow + (size_t)m0 * In + 32, In);

    const int NCH = KC / 64;  // 40
#pragma unroll 2
    for (int c = 0; c < NCH - 1; ++c) {
      const int cur = c & 1;
      {  // async-prefetch next 64x64 B tile into the other buffer (always valid here)
        int row = tid >> 3, col = (tid & 7) * 8;
        const unsigned short* s0 = wcomb + (size_t)((c + 1) * 64 + row) * Hn + nb + col;
        unsigned short* l0 = BtR + (cur ^ 1) * (64 * 72) + row * 72 + col;
        stage16(s0, l0);
        stage16(s0 + (size_t)32 * Hn, l0 + 32 * 72);
      }
      // register prefetch of next chunk's A fragments (x-part or h-part, never straddles)
      const int k0 = (c + 1) * 64;
      const unsigned short* p0;
      size_t st0;
      if (k0 < In) { p0 = xrow   + (size_t)m0 * In + (size_t)k0;        st0 = In; }
      else         { p0 = hb_cur + (size_t)m0 * Hn + (size_t)(k0 - In); st0 = Hn; }
      v16bf aN0 = load_a_frag(p0, st0);
      const int k1 = k0 + 32;
      const unsigned short* p1;
      size_t st1;
      if (k1 < In) { p1 = xrow   + (size_t)m0 * In + (size_t)k1;        st1 = In; }
      else         { p1 = hb_cur + (size_t)m0 * Hn + (size_t)(k1 - In); st1 = Hn; }
      v16bf aN1 = load_a_frag(p1, st1);

      const unsigned short* bt  = BtR + cur * (64 * 72);        // K rows 0..31
      const unsigned short* bt2 = bt + 32 * 72;                 // K rows 32..63
      v16bf b;
      b = load_b_frag(bt, 72, 0);   a0 = WMMA_BF16(aF0, b, a0);
      b = load_b_frag(bt, 72, 16);  a1 = WMMA_BF16(aF0, b, a1);
      b = load_b_frag(bt, 72, 32);  a2 = WMMA_BF16(aF0, b, a2);
      b = load_b_frag(bt, 72, 48);  a3 = WMMA_BF16(aF0, b, a3);
      b = load_b_frag(bt2, 72, 0);  a0 = WMMA_BF16(aF1, b, a0);
      b = load_b_frag(bt2, 72, 16); a1 = WMMA_BF16(aF1, b, a1);
      b = load_b_frag(bt2, 72, 32); a2 = WMMA_BF16(aF1, b, a2);
      b = load_b_frag(bt2, 72, 48); a3 = WMMA_BF16(aF1, b, a3);

      aF0 = aN0;
      aF1 = aN1;
      wait_async0();
      __syncthreads();
    }
    {  // peeled last chunk: buffer parity (NCH-1)&1
      const unsigned short* bt  = BtR + ((NCH - 1) & 1) * (64 * 72);
      const unsigned short* bt2 = bt + 32 * 72;
      v16bf b;
      b = load_b_frag(bt, 72, 0);   a0 = WMMA_BF16(aF0, b, a0);
      b = load_b_frag(bt, 72, 16);  a1 = WMMA_BF16(aF0, b, a1);
      b = load_b_frag(bt, 72, 32);  a2 = WMMA_BF16(aF0, b, a2);
      b = load_b_frag(bt, 72, 48);  a3 = WMMA_BF16(aF0, b, a3);
      b = load_b_frag(bt2, 72, 0);  a0 = WMMA_BF16(aF1, b, a0);
      b = load_b_frag(bt2, 72, 16); a1 = WMMA_BF16(aF1, b, a1);
      b = load_b_frag(bt2, 72, 32); a2 = WMMA_BF16(aF1, b, a2);
      b = load_b_frag(bt2, 72, 48); a3 = WMMA_BF16(aF1, b, a3);
      __syncthreads();
    }

    // ---- gate epilogue: h_{t+1} = (1-s)h_t + s*tanh(u + bias_c) ----
    const int mb = m0 + 8 * (lane >> 4);
#pragma unroll
    for (int q = 0; q < 4; ++q) {
      const int n = nb + q * 16 + (lane & 15);
      const float bc = bias_c[n];
      v8f& acc = (q == 0) ? a0 : (q == 1) ? a1 : (q == 2) ? a2 : a3;
#pragma unroll
      for (int r = 0; r < 8; ++r) {
        const size_t idx = (size_t)(mb + r) * Hn + n;
        const float u  = acc[r] + bc;
        const float hn = omg * hf_cur[idx] + sg * tanhf(u);
        hf_nxt[idx] = hn;
        hb_nxt[idx] = f2bf(hn);
      }
    }

    // ---- device-wide barrier for this step ----
    __syncthreads();
    if (tid == 0) {
      __hip_atomic_fetch_add(&bar[t], 1u, __ATOMIC_ACQ_REL, __HIP_MEMORY_SCOPE_AGENT);
      while (__hip_atomic_load(&bar[t], __ATOMIC_ACQUIRE, __HIP_MEMORY_SCOPE_AGENT) <
             (unsigned)gridDim.x) {
        __builtin_amdgcn_s_sleep(1);
      }
    }
    __syncthreads();
  }

  // y_{T-1} = h_T @ dec_w^T + dec_b   (h_T lives at parity 0 since T is even)
  dec_step(Tn - 1, hb, wdec, dec_b, out, BtD);

  // hidden output: copy h_T (fp32, parity 0)
  float* outH = out + (size_t)Tn * Bsz * On;
  const float* hfin = hf;
  const size_t base = (size_t)blockIdx.x * 256 + tid;
#pragma unroll 1
  for (int j = 0; j < (Bsz * Hn) / (NBLK * 256); ++j)
    outH[base + (size_t)j * (NBLK * 256)] = hfin[base + (size_t)j * (NBLK * 256)];
}

// ---------------- prep kernels (run once per call, all deterministic) ----------------

__global__ void k_zero_bar(unsigned int* bar) {
  int i = blockIdx.x * 256 + threadIdx.x;
  if (i < Tn) bar[i] = 0u;
}

__global__ void k_bias(const float* enc_b, const float* rec_b, const float* a, float* bias_c) {
  int i = blockIdx.x * 256 + threadIdx.x;
  bias_c[i] = enc_b[i] + a[0] * rec_b[i];
}

__global__ void k_h0(const float* hi_w, const float* hi_b, float* hf, unsigned short* hb) {
  int idx = blockIdx.x * 256 + threadIdx.x;
  int h = idx % Hn;
  float v = hi_w[h] + hi_b[h];
  hf[idx] = v;
  hb[idx] = f2bf(v);
}

__global__ void k_xb(const float* __restrict__ x, unsigned short* __restrict__ xb) {
  size_t o = ((size_t)blockIdx.x * 256 + threadIdx.x) * 8;
  int    i  = (int)(o % In);
  size_t tb = o / In;
  int    tt = (int)(tb / Bsz);
  int    bb = (int)(tb % Bsz);
  const float* src = x + ((size_t)bb * Tn + tt) * In + i;   // [B,T,I] -> time-major bf16
  float4 f0 = *(const float4*)src;
  float4 f1 = *(const float4*)(src + 4);
  union { unsigned short s[8]; uint4 q; } d;
  d.s[0] = f2bf(f0.x); d.s[1] = f2bf(f0.y); d.s[2] = f2bf(f0.z); d.s[3] = f2bf(f0.w);
  d.s[4] = f2bf(f1.x); d.s[5] = f2bf(f1.y); d.s[6] = f2bf(f1.z); d.s[7] = f2bf(f1.w);
  *(uint4*)(xb + o) = d.q;
}

__global__ void k_wcomb(const float* __restrict__ enc_w, const float* __restrict__ rec_w,
                        const float* a, unsigned short* __restrict__ wcomb) {
  size_t idx = (size_t)blockIdx.x * 256 + threadIdx.x;   // [KC, H] layout
  int k = (int)(idx / Hn), n = (int)(idx % Hn);
  float v = (k < In) ? enc_w[(size_t)n * In + k]
                     : a[0] * rec_w[(size_t)n * Hn + (size_t)(k - In)];
  wcomb[idx] = f2bf(v);
}

__global__ void k_wdec(const float* __restrict__ dec_w, unsigned short* __restrict__ wdec) {
  size_t idx = (size_t)blockIdx.x * 256 + threadIdx.x;   // [H, O] layout
  int k = (int)(idx / On), n = (int)(idx % On);
  wdec[idx] = f2bf(dec_w[(size_t)n * Hn + k]);
}

// ---------------- launch ----------------

extern "C" void kernel_launch(void* const* d_in, const int* in_sizes, int n_in,
                              void* d_out, int out_size, void* d_ws, size_t ws_size,
                              hipStream_t stream) {
  (void)in_sizes; (void)n_in; (void)out_size; (void)ws_size;
  const float* x     = (const float*)d_in[0];
  const float* dt    = (const float*)d_in[1];
  const float* a     = (const float*)d_in[2];
  const float* enc_w = (const float*)d_in[3];
  const float* enc_b = (const float*)d_in[4];
  const float* rec_w = (const float*)d_in[5];
  const float* rec_b = (const float*)d_in[6];
  const float* dec_w = (const float*)d_in[7];
  const float* dec_b = (const float*)d_in[8];
  const float* hi_w  = (const float*)d_in[9];
  const float* hi_b  = (const float*)d_in[10];
  float* out = (float*)d_out;

  char* ws = (char*)d_ws;
  size_t off = 0;
  auto alloc = [&](size_t bytes) -> void* {
    void* p = ws + off;
    off = (off + bytes + 255) & ~(size_t)255;
    return p;
  };
  unsigned short* xb    = (unsigned short*)alloc((size_t)Tn * Bsz * In * 2);  // 64 MB
  unsigned short* wcomb = (unsigned short*)alloc((size_t)KC * Hn * 2);        // 10 MB
  unsigned short* wdec  = (unsigned short*)alloc((size_t)Hn * On * 2);        // 2 MB
  float*          biasc = (float*)alloc((size_t)Hn * 4);
  float*          hf    = (float*)alloc((size_t)2 * Bsz * Hn * 4);            // 2 MB
  unsigned short* hb    = (unsigned short*)alloc((size_t)2 * Bsz * Hn * 2);   // 1 MB
  unsigned int*   bar   = (unsigned int*)alloc((size_t)Tn * 4);

  k_zero_bar<<<2, 256, 0, stream>>>(bar);
  k_bias<<<Hn / 256, 256, 0, stream>>>(enc_b, rec_b, a, biasc);
  k_h0<<<(Bsz * Hn) / 256, 256, 0, stream>>>(hi_w, hi_b, hf, hb);
  k_xb<<<((size_t)Tn * Bsz * In / 8) / 256, 256, 0, stream>>>(x, xb);
  k_wcomb<<<((size_t)KC * Hn) / 256, 256, 0, stream>>>(enc_w, rec_w, a, wcomb);
  k_wdec<<<((size_t)Hn * On) / 256, 256, 0, stream>>>(dec_w, wdec);

  rnn_persistent<<<NBLK, 256, 0, stream>>>(xb, wcomb, wdec, biasc, dec_b, dt,
                                           hf, hb, out, bar);
}
